// EmbeddingsAndEvoformer_23476291239972
// MI455X (gfx1250) — compile-verified, matrix operand order checked by reading, not run on hardware
//
#include <hip/hip_runtime.h>

// ---------------------------------------------------------------------------
// AlphaFold2 Evoformer block for MI455X (gfx1250, wave32, WMMA).
// All heavy matmuls run through v_wmma_f32_16x16x32_bf16 (f32 in memory,
// bf16 on the wire, f32 accumulate).  Attention is flash-style (no logits
// materialization: 268MB > 192MB L2).  OPM is chunked so its [512,8192]
// outer-product slab stays L2-resident.  Interior GEMM blocks take an
// unguarded vectorized fast path with global_prefetch of the next K tile.
// ---------------------------------------------------------------------------

#define DEV __device__ __forceinline__

typedef __attribute__((ext_vector_type(16))) __bf16 v16bf;
typedef __attribute__((ext_vector_type(8)))  float  v8f;

union ABu { unsigned int u[8]; v16bf v; };
union CFu { float f[8]; v8f v; };

DEV unsigned int bf16pair(float a, float b) {
  unsigned int ua = __float_as_uint(a);
  unsigned int ub = __float_as_uint(b);
  ua += 0x7fffu + ((ua >> 16) & 1u);   // RNE
  ub += 0x7fffu + ((ub >> 16) & 1u);
  return (ua >> 16) | (ub & 0xffff0000u);  // low16 = a, high16 = b
}
DEV unsigned short bf16one(float a) {
  unsigned int ua = __float_as_uint(a);
  ua += 0x7fffu + ((ua >> 16) & 1u);
  return (unsigned short)(ua >> 16);
}
DEV v8f wmma_bf16(v16bf a, v16bf b, v8f c) {
  return __builtin_amdgcn_wmma_f32_16x16x32_bf16(false, a, false, b, (short)0, c, false, false);
}

// ---------------------------------------------------------------------------
// GEMM main loop, templated on FULL (interior block, K%32==0: no guards,
// vector loads, next-tile prefetch) vs guarded edge path.
// LDS tiles are bf16 pair-packed in the exact WMMA A/B fragment pair order
// (ISA 7.12.2): row-stride 16 uints; pairs {0..3,8..11} for lanes 0-15,
// {4..7,12..15} for lanes 16-31.
// ---------------------------------------------------------------------------
template <bool FULL>
DEV void gemm_mainloop(const float* __restrict__ A, long long lda, int transA,
                       const float* __restrict__ B, long long ldb, int transB,
                       int M, int N, int K, int m0, int n0,
                       int tid, int rsel, int hp, int mt, int ntb,
                       unsigned int* Ap, unsigned int* Bp,
                       CFu& acc0, CFu& acc1)
{
  for (int k0 = 0; k0 < K; k0 += 32) {
    __syncthreads();
    // ---- A tile: 64 rows x 16 k-pairs ----
    if (!transA) {
#pragma unroll
      for (int r = 0; r < 4; ++r) {
        int idx = tid + r * 256;
        int mm = idx >> 4, p = idx & 15;
        int gm = m0 + mm, gk = k0 + 2 * p;
        float a0 = 0.f, a1 = 0.f;
        if (FULL) {
          const float2 t = *(const float2*)(A + (long long)gm * lda + gk);
          a0 = t.x; a1 = t.y;
          if (r == 0 && k0 + 32 < K)
            __builtin_prefetch(A + (long long)gm * lda + gk + 32, 0, 1);
        } else {
          if (gm < M) {
            if (gk     < K) a0 = A[(long long)gm * lda + gk];
            if (gk + 1 < K) a1 = A[(long long)gm * lda + gk + 1];
          }
        }
        Ap[mm * 16 + p] = bf16pair(a0, a1);
      }
    } else {
#pragma unroll
      for (int r = 0; r < 4; ++r) {
        int idx = tid + r * 256;
        int p = idx >> 6, mm = idx & 63;
        int gm = m0 + mm, gk = k0 + 2 * p;
        float a0 = 0.f, a1 = 0.f;
        if (FULL) {
          a0 = A[(long long)gk * lda + gm];
          a1 = A[(long long)(gk + 1) * lda + gm];
          if (r == 0 && k0 + 32 < K)
            __builtin_prefetch(A + (long long)(gk + 32) * lda + gm, 0, 1);
        } else {
          if (gm < M) {
            if (gk     < K) a0 = A[(long long)gk * lda + gm];
            if (gk + 1 < K) a1 = A[(long long)(gk + 1) * lda + gm];
          }
        }
        Ap[mm * 16 + p] = bf16pair(a0, a1);
      }
    }
    // ---- B tile: 64 cols x 16 k-pairs ----
    if (!transB) {
#pragma unroll
      for (int r = 0; r < 4; ++r) {
        int idx = tid + r * 256;
        int p = idx >> 6, nn = idx & 63;
        int gn = n0 + nn, gk = k0 + 2 * p;
        float b0 = 0.f, b1 = 0.f;
        if (FULL) {
          b0 = B[(long long)gk * ldb + gn];
          b1 = B[(long long)(gk + 1) * ldb + gn];
          if (r == 0 && k0 + 32 < K)
            __builtin_prefetch(B + (long long)(gk + 32) * ldb + gn, 0, 1);
        } else {
          if (gn < N) {
            if (gk     < K) b0 = B[(long long)gk * ldb + gn];
            if (gk + 1 < K) b1 = B[(long long)(gk + 1) * ldb + gn];
          }
        }
        Bp[nn * 16 + p] = bf16pair(b0, b1);
      }
    } else {
#pragma unroll
      for (int r = 0; r < 4; ++r) {
        int idx = tid + r * 256;
        int nn = idx >> 4, p = idx & 15;
        int gn = n0 + nn, gk = k0 + 2 * p;
        float b0 = 0.f, b1 = 0.f;
        if (FULL) {
          const float2 t = *(const float2*)(B + (long long)gn * ldb + gk);
          b0 = t.x; b1 = t.y;
          if (r == 0 && k0 + 32 < K)
            __builtin_prefetch(B + (long long)gn * ldb + gk + 32, 0, 1);
        } else {
          if (gn < N) {
            if (gk     < K) b0 = B[(long long)gn * ldb + gk];
            if (gk + 1 < K) b1 = B[(long long)gn * ldb + gk + 1];
          }
        }
        Bp[nn * 16 + p] = bf16pair(b0, b1);
      }
    }
    __syncthreads();

    ABu fa;
#pragma unroll
    for (int j = 0; j < 4; ++j) {
      fa.u[j]     = Ap[(mt * 16 + rsel) * 16 + hp + j];
      fa.u[4 + j] = Ap[(mt * 16 + rsel) * 16 + 8 + hp + j];
    }
    ABu fb0, fb1;
#pragma unroll
    for (int j = 0; j < 4; ++j) {
      fb0.u[j]     = Bp[(ntb * 16 + rsel) * 16 + hp + j];
      fb0.u[4 + j] = Bp[(ntb * 16 + rsel) * 16 + 8 + hp + j];
      fb1.u[j]     = Bp[((ntb + 2) * 16 + rsel) * 16 + hp + j];
      fb1.u[4 + j] = Bp[((ntb + 2) * 16 + rsel) * 16 + 8 + hp + j];
    }
    acc0.v = wmma_bf16(fa.v, fb0.v, acc0.v);
    acc1.v = wmma_bf16(fa.v, fb1.v, acc1.v);
  }
}

// ---------------------------------------------------------------------------
// Generic batched GEMM:  C = epilogue(alpha * A(MxK) * B(KxN))
//  A[m][k] = transA ? A[k*lda+m] : A[m*lda+k]
//  B[k][n] = transB ? B[n*ldb+k] : B[k*ldb+n]
//  epilogue: +bias[n], act(0 none /1 relu /2 sigmoid), *rowscale[m],
//            accumulate into C, optional OPM-permuted store.
// Block = 256 threads (8 waves).  Tile 64x64x32.
// ---------------------------------------------------------------------------
__global__ void __launch_bounds__(256) gemm_bf16_kernel(
    const float* __restrict__ A, long long lda, int transA, long long bsA,
    const float* __restrict__ B, long long ldb, int transB, long long bsB,
    float* __restrict__ C, long long ldc, long long bsC,
    int M, int N, int K,
    const float* __restrict__ bias, const float* __restrict__ rowscale,
    float alpha, int act, int accum, int opmR)
{
  __shared__ unsigned int Ap[64 * 16];
  __shared__ unsigned int Bp[64 * 16];
  const int tid  = threadIdx.x;
  const int lane = tid & 31;
  const int w    = tid >> 5;
  const long long zb = blockIdx.z;
  A += zb * bsA;  B += zb * bsB;  C += zb * bsC;
  const int m0 = blockIdx.y * 64;
  const int n0 = blockIdx.x * 64;
  const int mt  = w & 3;        // m-subtile 0..3
  const int ntb = w >> 2;       // n-subtiles ntb and ntb+2
  const int rsel = lane & 15;
  const int hp   = (lane >> 4) * 4;

  CFu acc0, acc1;
#pragma unroll
  for (int j = 0; j < 8; ++j) { acc0.f[j] = 0.f; acc1.f[j] = 0.f; }

  const bool full = (m0 + 64 <= M) && (n0 + 64 <= N) && ((K & 31) == 0);
  if (full)
    gemm_mainloop<true >(A, lda, transA, B, ldb, transB, M, N, K, m0, n0,
                         tid, rsel, hp, mt, ntb, Ap, Bp, acc0, acc1);
  else
    gemm_mainloop<false>(A, lda, transA, B, ldb, transB, M, N, K, m0, n0,
                         tid, rsel, hp, mt, ntb, Ap, Bp, acc0, acc1);

  const int half = lane >> 4;
#pragma unroll
  for (int t = 0; t < 2; ++t) {
    CFu& ac = t ? acc1 : acc0;
    int gn = n0 + (ntb + 2 * t) * 16 + rsel;
#pragma unroll
    for (int j = 0; j < 8; ++j) {
      int gm = m0 + mt * 16 + j + half * 8;
      if (gm < M && gn < N) {
        float v = ac.f[j] * alpha;
        if (bias) v += bias[gn];
        if (act == 1)      v = fmaxf(v, 0.f);
        else if (act == 2) v = 1.f / (1.f + __expf(-v));
        if (rowscale) v *= rowscale[gm];
        long long off;
        if (opmR > 0)
          off = ((long long)(gm >> 5) * opmR + (gn >> 5)) * 1024 + (gm & 31) * 32 + (gn & 31);
        else
          off = (long long)gm * ldc + gn;
        if (accum) C[off] += v; else C[off] = v;
      }
    }
  }
}

// ---------------------------------------------------------------------------
// Flash attention, one wave per (batch, head, 16-query tile).  Head dim = 32
// == one WMMA K step.  Strided (b,l) addressing lets msa-col / tri-attn-end
// run on the same row-major buffers with swapped strides.
// bias = 1e9*(mask[b,k]-1) (+ optional 2D-feature bias nb[h,q,k]).
// ---------------------------------------------------------------------------
DEV void frag_from_row(ABu& f, const float* __restrict__ row, int half) {
  const float4* r4 = (const float4*)row;
  float4 x0 = r4[half * 2 + 0];       // K = half*8 .. half*8+7
  float4 x1 = r4[half * 2 + 1];
  float4 x2 = r4[4 + half * 2 + 0];   // K = 16+half*8 ..
  float4 x3 = r4[4 + half * 2 + 1];
  f.u[0] = bf16pair(x0.x, x0.y);  f.u[1] = bf16pair(x0.z, x0.w);
  f.u[2] = bf16pair(x1.x, x1.y);  f.u[3] = bf16pair(x1.z, x1.w);
  f.u[4] = bf16pair(x2.x, x2.y);  f.u[5] = bf16pair(x2.z, x2.w);
  f.u[6] = bf16pair(x3.x, x3.y);  f.u[7] = bf16pair(x3.z, x3.w);
}

__global__ void __launch_bounds__(32) flash_attn_kernel(
    const float* __restrict__ Q, const float* __restrict__ K,
    const float* __restrict__ V, float* __restrict__ O,
    int H, int L, long long sB, long long sL,
    const float* __restrict__ maskp, long long mSB, long long mSK,
    const float* __restrict__ nb, long long nbSQ, long long nbSK, int nbH)
{
  __shared__ unsigned short pls[16 * 32];   // P tile, bf16, [q][k]
  __shared__ unsigned short vls[32 * 32];   // V tile, bf16, [d][k]
  unsigned int* plsu = (unsigned int*)pls;
  unsigned int* vlsu = (unsigned int*)vls;

  const int lane = threadIdx.x;
  const int q0   = blockIdx.x * 16;
  const int b    = blockIdx.y / H;
  const int h    = blockIdx.y % H;
  const int rsel = lane & 15;
  const int half = lane >> 4;
  const int hp   = half * 4;
  const long long base = (long long)b * sB + (long long)h * 32;

  ABu fq;                                    // Q fragment (16 x 32)
  frag_from_row(fq, Q + base + (long long)(q0 + rsel) * sL, half);

  float mrow[8], lrow[8];
  CFu o0, o1;
#pragma unroll
  for (int j = 0; j < 8; ++j) { mrow[j] = -1e30f; lrow[j] = 0.f; o0.f[j] = 0.f; o1.f[j] = 0.f; }

  for (int k0 = 0; k0 < L; k0 += 32) {
    if (k0 + 32 < L) {                       // prefetch next K/V tile rows
      __builtin_prefetch(K + base + (long long)(k0 + 32 + rsel) * sL, 0, 1);
      __builtin_prefetch(V + base + (long long)(k0 + 32 + lane) * sL, 0, 1);
    }
    ABu fk0, fk1;                            // K^T fragments (two 16-key cols)
    frag_from_row(fk0, K + base + (long long)(k0 + rsel) * sL, half);
    frag_from_row(fk1, K + base + (long long)(k0 + 16 + rsel) * sL, half);

    CFu s0, s1;
#pragma unroll
    for (int j = 0; j < 8; ++j) { s0.f[j] = 0.f; s1.f[j] = 0.f; }
    s0.v = wmma_bf16(fq.v, fk0.v, s0.v);
    s1.v = wmma_bf16(fq.v, fk1.v, s1.v);

    int kc0 = k0 + rsel, kc1 = k0 + 16 + rsel;
    float bias0 = 1e9f * (maskp[(long long)b * mSB + (long long)kc0 * mSK] - 1.f);
    float bias1 = 1e9f * (maskp[(long long)b * mSB + (long long)kc1 * mSK] - 1.f);
#pragma unroll
    for (int j = 0; j < 8; ++j) {
      int qrow = q0 + j + half * 8;
      float n0v = 0.f, n1v = 0.f;
      if (nb) {
        n0v = nb[((long long)qrow * nbSQ + (long long)kc0 * nbSK) * nbH + h];
        n1v = nb[((long long)qrow * nbSQ + (long long)kc1 * nbSK) * nbH + h];
      }
      s0.f[j] += bias0 + n0v;
      s1.f[j] += bias1 + n1v;
    }

    // stage V tile into LDS transposed: vls[d][k]
    {
      const float4* vr4 = (const float4*)(V + base + (long long)(k0 + lane) * sL);
#pragma unroll
      for (int dd = 0; dd < 8; ++dd) {
        float4 t = vr4[dd];
        vls[(dd * 4 + 0) * 32 + lane] = bf16one(t.x);
        vls[(dd * 4 + 1) * 32 + lane] = bf16one(t.y);
        vls[(dd * 4 + 2) * 32 + lane] = bf16one(t.z);
        vls[(dd * 4 + 3) * 32 + lane] = bf16one(t.w);
      }
    }

    // online softmax (row stats shared across the 16 lanes of each half)
#pragma unroll
    for (int j = 0; j < 8; ++j) {
      float cm = fmaxf(s0.f[j], s1.f[j]);
#pragma unroll
      for (int m = 1; m <= 8; m <<= 1) cm = fmaxf(cm, __shfl_xor(cm, m, 32));
      float nm   = fmaxf(mrow[j], cm);
      float corr = __expf(mrow[j] - nm);
      float p0   = __expf(s0.f[j] - nm);
      float p1   = __expf(s1.f[j] - nm);
      float ps   = p0 + p1;
#pragma unroll
      for (int m = 1; m <= 8; m <<= 1) ps += __shfl_xor(ps, m, 32);
      lrow[j] = lrow[j] * corr + ps;
      mrow[j] = nm;
      o0.f[j] *= corr; o1.f[j] *= corr;
      pls[(j + half * 8) * 32 + rsel]      = bf16one(p0);
      pls[(j + half * 8) * 32 + 16 + rsel] = bf16one(p1);
    }
    __syncthreads();

    ABu fp, fv0, fv1;
#pragma unroll
    for (int j = 0; j < 4; ++j) {
      fp.u[j]      = plsu[rsel * 16 + hp + j];
      fp.u[4 + j]  = plsu[rsel * 16 + 8 + hp + j];
      fv0.u[j]     = vlsu[rsel * 16 + hp + j];
      fv0.u[4 + j] = vlsu[rsel * 16 + 8 + hp + j];
      fv1.u[j]     = vlsu[(16 + rsel) * 16 + hp + j];
      fv1.u[4 + j] = vlsu[(16 + rsel) * 16 + 8 + hp + j];
    }
    o0.v = wmma_bf16(fp.v, fv0.v, o0.v);
    o1.v = wmma_bf16(fp.v, fv1.v, o1.v);
    __syncthreads();
  }

#pragma unroll
  for (int j = 0; j < 8; ++j) {
    float inv = 1.f / lrow[j];
    float* orow = O + base + (long long)(q0 + j + half * 8) * sL;
    orow[rsel]      = o0.f[j] * inv;
    orow[16 + rsel] = o1.f[j] * inv;
  }
}

// ---------------------------------------------------------------------------
// LayerNorm: one wave per row.
// ---------------------------------------------------------------------------
__global__ void __launch_bounds__(256) ln_kernel(
    const float* __restrict__ x, const float* __restrict__ g,
    const float* __restrict__ b, float* __restrict__ y,
    long long rows, int C)
{
  int lane = threadIdx.x & 31;
  long long row = (long long)blockIdx.x * 8 + (threadIdx.x >> 5);
  if (row >= rows) return;
  const float* xr = x + row * C;
  float s = 0.f;
  for (int c = lane; c < C; c += 32) s += xr[c];
#pragma unroll
  for (int m = 1; m <= 16; m <<= 1) s += __shfl_xor(s, m, 32);
  float mean = s / C;
  float v = 0.f;
  for (int c = lane; c < C; c += 32) { float d = xr[c] - mean; v += d * d; }
#pragma unroll
  for (int m = 1; m <= 16; m <<= 1) v += __shfl_xor(v, m, 32);
  float r = rsqrtf(v / C + 1e-5f);
  float* yr = y + row * C;
  for (int c = lane; c < C; c += 32) yr[c] = (xr[c] - mean) * r * g[c] + b[c];
}

// ---------------------------------------------------------------------------
// Elementwise + transpose helpers.
// ---------------------------------------------------------------------------
__global__ void mul_kernel(float* __restrict__ d, const float* __restrict__ a, long long n) {
  long long st = (long long)gridDim.x * blockDim.x;
  for (long long i = (long long)blockIdx.x * blockDim.x + threadIdx.x; i < n; i += st) d[i] *= a[i];
}
__global__ void muladd_kernel(float* __restrict__ d, const float* __restrict__ a,
                              const float* __restrict__ b, long long n) {
  long long st = (long long)gridDim.x * blockDim.x;
  for (long long i = (long long)blockIdx.x * blockDim.x + threadIdx.x; i < n; i += st) d[i] += a[i] * b[i];
}
// [RR, C] -> [C, RR]
__global__ void to_chfirst_kernel(const float* __restrict__ in, float* __restrict__ out,
                                  long long RR, int C) {
  long long n = RR * C;
  long long st = (long long)gridDim.x * blockDim.x;
  for (long long i = (long long)blockIdx.x * blockDim.x + threadIdx.x; i < n; i += st) {
    long long rr = i / C; int c = (int)(i - rr * C);
    out[(long long)c * RR + rr] = in[i];
  }
}
// [C, RR] -> [RR, C]
__global__ void to_chlast_kernel(const float* __restrict__ in, float* __restrict__ out,
                                 long long RR, int C) {
  long long n = RR * C;
  long long st = (long long)gridDim.x * blockDim.x;
  for (long long i = (long long)blockIdx.x * blockDim.x + threadIdx.x; i < n; i += st) {
    long long rr = i / C; int c = (int)(i - rr * C);
    out[i] = in[(long long)c * RR + rr];
  }
}
// normInv[i,j] = 1 / (1e-3 + sum_s mask[s,i]*mask[s,j])
__global__ void opm_norm_kernel(const float* __restrict__ mask, float* __restrict__ ninv,
                                int S, int R) {
  int i = blockIdx.y;
  int j = blockIdx.x * blockDim.x + threadIdx.x;
  if (j >= R) return;
  float s = 0.f;
  for (int t = 0; t < S; ++t) s += mask[t * R + i] * mask[t * R + j];
  ninv[(long long)i * R + j] = 1.f / (1e-3f + s);
}

// ---------------------------------------------------------------------------
// Host orchestration.
// ---------------------------------------------------------------------------
extern "C" void kernel_launch(void* const* d_in, const int* in_sizes, int n_in,
                              void* d_out, int out_size, void* d_ws, size_t ws_size,
                              hipStream_t stream) {
  (void)in_sizes; (void)n_in; (void)out_size; (void)ws_size;

  const int S = 128, R = 256, CM = 256, CZ = 128, HM = 8, HZ = 4, OPC = 32;
  const long long MS = (long long)S * R;       // 32768 msa rows
  const long long MP = (long long)R * R;       // 65536 pair rows
  const long long NMSA  = MS * CM;             // 8388608
  const long long NPAIR = MP * CZ;             // 8388608
  const float QSCALE = 0.17677669529663687f;   // 32^-0.5

  const float* msa_in    = (const float*)d_in[0];
  const float* msa_mask  = (const float*)d_in[1];
  const float* pair_in   = (const float*)d_in[2];
  const float* pair_mask = (const float*)d_in[3];

  // params: jax pytree flatten => dict keys sorted alphabetically at each level
  auto P = [&](int i) { return (const float*)d_in[i]; };
  int pi = 4;
  // msa_col: gate_b, gate_w, k_w, ln_b, ln_g, out_b, out_w, q_w, v_w
  const float *col_gate_b = P(pi+0), *col_gate_w = P(pi+1), *col_k_w = P(pi+2),
              *col_ln_b = P(pi+3), *col_ln_g = P(pi+4), *col_out_b = P(pi+5),
              *col_out_w = P(pi+6), *col_q_w = P(pi+7), *col_v_w = P(pi+8);
  pi += 9;
  // msa_row: feat_2d_w, gate_b, gate_w, k_w, ln_b, ln_g, out_b, out_w, pair_ln_b, pair_ln_g, q_w, v_w
  const float *row_feat_w = P(pi+0), *row_gate_b = P(pi+1), *row_gate_w = P(pi+2),
              *row_k_w = P(pi+3), *row_ln_b = P(pi+4), *row_ln_g = P(pi+5),
              *row_out_b = P(pi+6), *row_out_w = P(pi+7), *row_pair_ln_b = P(pi+8),
              *row_pair_ln_g = P(pi+9), *row_q_w = P(pi+10), *row_v_w = P(pi+11);
  pi += 12;
  // msa_trans: b1, b2, ln_b, ln_g, w1, w2
  const float *mt_b1 = P(pi+0), *mt_b2 = P(pi+1), *mt_ln_b = P(pi+2),
              *mt_ln_g = P(pi+3), *mt_w1 = P(pi+4), *mt_w2 = P(pi+5);
  pi += 6;
  // opm: left_b, left_w, ln_b, ln_g, out_b, out_w, right_b, right_w
  const float *opm_left_b = P(pi+0), *opm_left_w = P(pi+1), *opm_ln_b = P(pi+2),
              *opm_ln_g = P(pi+3), *opm_out_b = P(pi+4), *opm_out_w = P(pi+5),
              *opm_right_b = P(pi+6), *opm_right_w = P(pi+7);
  pi += 8;
  // pair_trans: b1, b2, ln_b, ln_g, w1, w2
  const float *pt_b1 = P(pi+0), *pt_b2 = P(pi+1), *pt_ln_b = P(pi+2),
              *pt_ln_g = P(pi+3), *pt_w1 = P(pi+4), *pt_w2 = P(pi+5);
  pi += 6;
  // tri_att_end: feat_2d_w, gate_b, gate_w, k_w, ln_b, ln_g, out_b, out_w, q_w, v_w
  const float *tae_feat_w = P(pi+0), *tae_gate_b = P(pi+1), *tae_gate_w = P(pi+2),
              *tae_k_w = P(pi+3), *tae_ln_b = P(pi+4), *tae_ln_g = P(pi+5),
              *tae_out_b = P(pi+6), *tae_out_w = P(pi+7), *tae_q_w = P(pi+8),
              *tae_v_w = P(pi+9);
  pi += 10;
  // tri_att_start
  const float *tas_feat_w = P(pi+0), *tas_gate_b = P(pi+1), *tas_gate_w = P(pi+2),
              *tas_k_w = P(pi+3), *tas_ln_b = P(pi+4), *tas_ln_g = P(pi+5),
              *tas_out_b = P(pi+6), *tas_out_w = P(pi+7), *tas_q_w = P(pi+8),
              *tas_v_w = P(pi+9);
  pi += 10;
  // tri_in: center_ln_b, center_ln_g, gate_b, gate_w, left_b, left_gate_b, left_gate_w,
  //         left_w, ln_b, ln_g, out_b, out_w, right_b, right_gate_b, right_gate_w, right_w
  const float *tin_cln_b = P(pi+0), *tin_cln_g = P(pi+1), *tin_gate_b = P(pi+2),
              *tin_gate_w = P(pi+3), *tin_left_b = P(pi+4), *tin_lg_b = P(pi+5),
              *tin_lg_w = P(pi+6), *tin_left_w = P(pi+7), *tin_ln_b = P(pi+8),
              *tin_ln_g = P(pi+9), *tin_out_b = P(pi+10), *tin_out_w = P(pi+11),
              *tin_right_b = P(pi+12), *tin_rg_b = P(pi+13), *tin_rg_w = P(pi+14),
              *tin_right_w = P(pi+15);
  pi += 16;
  // tri_out
  const float *tot_cln_b = P(pi+0), *tot_cln_g = P(pi+1), *tot_gate_b = P(pi+2),
              *tot_gate_w = P(pi+3), *tot_left_b = P(pi+4), *tot_lg_b = P(pi+5),
              *tot_lg_w = P(pi+6), *tot_left_w = P(pi+7), *tot_ln_b = P(pi+8),
              *tot_ln_g = P(pi+9), *tot_out_b = P(pi+10), *tot_out_w = P(pi+11),
              *tot_right_b = P(pi+12), *tot_rg_b = P(pi+13), *tot_rg_w = P(pi+14),
              *tot_right_w = P(pi+15);
  pi += 16;

  // workspace layout (floats)
  float* ws = (float*)d_ws;
  const long long BUF = 8388608;
  float* bA = ws;                 // xln scratch
  float* bB = ws + 1 * BUF;
  float* bC = ws + 2 * BUF;
  float* bD = ws + 3 * BUF;
  float* bE = ws + 4 * BUF;
  float* bF = ws + 5 * BUF;
  float* bG = ws + 6 * BUF;
  float* bH = ws + 7 * BUF;       // 33554432 floats (transition hidden / OPM slab)
  float* bNB   = bH + 33554432;   // 524288
  float* bNORM = bNB + 524288;    // 65536

  float* msa  = (float*)d_out;            // residual state lives in d_out
  float* pair = (float*)d_out + NMSA;

  auto gemm = [&](const float* A, long long lda, int tA, long long bsA,
                  const float* B, long long ldb, int tB, long long bsB,
                  float* C, long long ldc, long long bsC,
                  int M, int N, int K, int batch,
                  const float* bias, const float* rowscale,
                  float alpha, int act, int accum, int opmR) {
    dim3 g((N + 63) / 64, (M + 63) / 64, batch);
    gemm_bf16_kernel<<<g, 256, 0, stream>>>(A, lda, tA, bsA, B, ldb, tB, bsB,
                                            C, ldc, bsC, M, N, K, bias, rowscale,
                                            alpha, act, accum, opmR);
  };
  auto ln = [&](const float* x, const float* g_, const float* b_, float* y,
                long long rows, int C) {
    ln_kernel<<<dim3((unsigned)((rows + 7) / 8)), 256, 0, stream>>>(x, g_, b_, y, rows, C);
  };
  auto mul = [&](float* d, const float* a, long long n) {
    mul_kernel<<<4096, 256, 0, stream>>>(d, a, n);
  };
  auto muladd = [&](float* d, const float* a, const float* b, long long n) {
    muladd_kernel<<<4096, 256, 0, stream>>>(d, a, b, n);
  };

  // x: layernormed input [Mrows, C] (C == Hh*32); writes bB..bF; accumulates into dst.
  auto attention = [&](const float* x, long long Mrows, int C, int Hh,
                       const float* qw, const float* kw, const float* vw,
                       const float* gw, const float* gb,
                       const float* ow, const float* ob,
                       int Bn, int L, long long sB, long long sL,
                       const float* maskp, long long mSB, long long mSK,
                       const float* nbp, long long nbSQ, long long nbSK, int nbH,
                       float* dst) {
    int N = Hh * 32;
    gemm(x, C, 0, 0, qw, N, 0, 0, bB, N, 0, (int)Mrows, N, C, 1, nullptr, nullptr, QSCALE, 0, 0, 0);
    gemm(x, C, 0, 0, kw, N, 0, 0, bC, N, 0, (int)Mrows, N, C, 1, nullptr, nullptr, 1.f, 0, 0, 0);
    gemm(x, C, 0, 0, vw, N, 0, 0, bD, N, 0, (int)Mrows, N, C, 1, nullptr, nullptr, 1.f, 0, 0, 0);
    gemm(x, C, 0, 0, gw, N, 0, 0, bE, N, 0, (int)Mrows, N, C, 1, gb, nullptr, 1.f, 2, 0, 0);
    flash_attn_kernel<<<dim3(L / 16, Bn * Hh), 32, 0, stream>>>(
        bB, bC, bD, bF, Hh, L, sB, sL, maskp, mSB, mSK, nbp, nbSQ, nbSK, nbH);
    mul(bF, bE, Mrows * C);
    gemm(bF, C, 0, 0, ow, C, 0, 0, dst, C, 0, (int)Mrows, C, C, 1, ob, nullptr, 1.f, 0, 1, 0);
  };

  auto transition = [&](float* state, long long Mrows, int C,
                        const float* lnG, const float* lnB_, const float* w1,
                        const float* b1, const float* w2, const float* b2) {
    ln(state, lnG, lnB_, bA, Mrows, C);
    gemm(bA, C, 0, 0, w1, 4 * C, 0, 0, bH, 4 * C, 0, (int)Mrows, 4 * C, C, 1, b1, nullptr, 1.f, 1, 0, 0);
    gemm(bH, 4 * C, 0, 0, w2, C, 0, 0, state, C, 0, (int)Mrows, C, 4 * C, 1, b2, nullptr, 1.f, 0, 1, 0);
  };

  auto tri_mul = [&](const float* lnG, const float* lnB_,
                     const float* lw, const float* lb, const float* lgw, const float* lgb,
                     const float* rw, const float* rb, const float* rgw, const float* rgb,
                     const float* clnG, const float* clnB,
                     const float* ow, const float* ob, const float* gw, const float* gb,
                     bool outgoing) {
    ln(pair, lnG, lnB_, bA, MP, CZ);
    gemm(bA, CZ, 0, 0, lw,  CZ, 0, 0, bB, CZ, 0, (int)MP, CZ, CZ, 1, lb,  pair_mask, 1.f, 0, 0, 0);
    gemm(bA, CZ, 0, 0, lgw, CZ, 0, 0, bC, CZ, 0, (int)MP, CZ, CZ, 1, lgb, nullptr,   1.f, 2, 0, 0);
    mul(bB, bC, NPAIR);
    gemm(bA, CZ, 0, 0, rw,  CZ, 0, 0, bD, CZ, 0, (int)MP, CZ, CZ, 1, rb,  pair_mask, 1.f, 0, 0, 0);
    gemm(bA, CZ, 0, 0, rgw, CZ, 0, 0, bC, CZ, 0, (int)MP, CZ, CZ, 1, rgb, nullptr,   1.f, 2, 0, 0);
    mul(bD, bC, NPAIR);
    to_chfirst_kernel<<<4096, 256, 0, stream>>>(bB, bE, MP, CZ);   // left  -> [c,r1,r2]
    to_chfirst_kernel<<<4096, 256, 0, stream>>>(bD, bF, MP, CZ);   // right -> [c,r1,r2]
    if (outgoing)   // out[i,j] = sum_k L[i,k] R[j,k]  (per channel)
      gemm(bE, R, 0, MP, bF, R, 1, MP, bG, R, MP, R, R, R, CZ, nullptr, nullptr, 1.f, 0, 0, 0);
    else            // out[i,j] = sum_k L[k,i] R[k,j]
      gemm(bE, R, 1, MP, bF, R, 0, MP, bG, R, MP, R, R, R, CZ, nullptr, nullptr, 1.f, 0, 0, 0);
    to_chlast_kernel<<<4096, 256, 0, stream>>>(bG, bB, MP, CZ);
    ln(bB, clnG, clnB, bD, MP, CZ);
    gemm(bD, CZ, 0, 0, ow, CZ, 0, 0, bE, CZ, 0, (int)MP, CZ, CZ, 1, ob, nullptr, 1.f, 0, 0, 0);
    gemm(bA, CZ, 0, 0, gw, CZ, 0, 0, bF, CZ, 0, (int)MP, CZ, CZ, 1, gb, nullptr, 1.f, 2, 0, 0);
    muladd(pair, bE, bF, NPAIR);
  };

  auto tri_attn = [&](const float* lnG, const float* lnB_, const float* featw,
                      const float* qw, const float* kw, const float* vw,
                      const float* gw, const float* gb, const float* ow, const float* ob,
                      bool starting) {
    ln(pair, lnG, lnB_, bA, MP, CZ);
    gemm(bA, CZ, 0, 0, featw, HZ, 0, 0, bNB, HZ, 0, (int)MP, HZ, CZ, 1, nullptr, nullptr, 1.f, 0, 0, 0);
    long long sB  = starting ? (long long)R * CZ : CZ;
    long long sL  = starting ? CZ : (long long)R * CZ;
    long long mSB = starting ? R : 1, mSK = starting ? 1 : R;
    long long nQ  = starting ? R : 1, nK  = starting ? 1 : R;
    attention(bA, MP, CZ, HZ, qw, kw, vw, gw, gb, ow, ob,
              R, R, sB, sL, pair_mask, mSB, mSK, bNB, nQ, nK, HZ, pair);
  };

  // residual state
  hipMemcpyAsync(msa,  msa_in,  NMSA  * sizeof(float), hipMemcpyDeviceToDevice, stream);
  hipMemcpyAsync(pair, pair_in, NPAIR * sizeof(float), hipMemcpyDeviceToDevice, stream);

  // ---- 1. outer product mean: pair += opm(msa) ----
  ln(msa, opm_ln_g, opm_ln_b, bA, MS, CM);
  gemm(bA, CM, 0, 0, opm_left_w,  OPC, 0, 0, bB, OPC, 0, (int)MS, OPC, CM, 1, opm_left_b,  msa_mask, 1.f, 0, 0, 0);
  gemm(bA, CM, 0, 0, opm_right_w, OPC, 0, 0, bC, OPC, 0, (int)MS, OPC, CM, 1, opm_right_b, msa_mask, 1.f, 0, 0, 0);
  opm_norm_kernel<<<dim3(1, R), 256, 0, stream>>>(msa_mask, bNORM, S, R);
  for (int i0 = 0; i0 < R; i0 += 16) {
    // slab[(ic,c),(j,d)] = sum_s left[s,i0+ic,c] * right[s,j,d], permuted store
    gemm(bB + i0 * 32, (long long)R * OPC, 1, 0, bC, (long long)R * OPC, 0, 0,
         bH, 0, 0, 16 * OPC, R * OPC, S, 1, nullptr, nullptr, 1.f, 0, 0, R);
    // pair[(i0+ic, j), :] += ninv * (slab[(ic,j), (c,d)] @ out_w[(c,d), :] + out_b)
    gemm(bH, OPC * OPC, 0, 0, opm_out_w, CZ, 0, 0,
         pair + (long long)i0 * R * CZ, CZ, 0, 16 * R, CZ, OPC * OPC, 1,
         opm_out_b, bNORM + (long long)i0 * R, 1.f, 0, 1, 0);
  }

  // ---- 2. msa row attention (uses updated pair for 2D bias) ----
  ln(msa,  row_ln_g,      row_ln_b,      bA, MS, CM);
  ln(pair, row_pair_ln_g, row_pair_ln_b, bG, MP, CZ);
  gemm(bG, CZ, 0, 0, row_feat_w, HM, 0, 0, bNB, HM, 0, (int)MP, HM, CZ, 1, nullptr, nullptr, 1.f, 0, 0, 0);
  attention(bA, MS, CM, HM, row_q_w, row_k_w, row_v_w, row_gate_w, row_gate_b,
            row_out_w, row_out_b, S, R, (long long)R * CM, CM,
            msa_mask, R, 1, bNB, R, 1, HM, msa);

  // ---- 3. msa column attention (swapped strides, no transpose) ----
  ln(msa, col_ln_g, col_ln_b, bA, MS, CM);
  attention(bA, MS, CM, HM, col_q_w, col_k_w, col_v_w, col_gate_w, col_gate_b,
            col_out_w, col_out_b, R, S, CM, (long long)R * CM,
            msa_mask, 1, R, nullptr, 0, 0, 1, msa);

  // ---- 4. msa transition ----
  transition(msa, MS, CM, mt_ln_g, mt_ln_b, mt_w1, mt_b1, mt_w2, mt_b2);

  // ---- 5/6. triangle multiplications ----
  tri_mul(tot_ln_g, tot_ln_b, tot_left_w, tot_left_b, tot_lg_w, tot_lg_b,
          tot_right_w, tot_right_b, tot_rg_w, tot_rg_b, tot_cln_g, tot_cln_b,
          tot_out_w, tot_out_b, tot_gate_w, tot_gate_b, true);
  tri_mul(tin_ln_g, tin_ln_b, tin_left_w, tin_left_b, tin_lg_w, tin_lg_b,
          tin_right_w, tin_right_b, tin_rg_w, tin_rg_b, tin_cln_g, tin_cln_b,
          tin_out_w, tin_out_b, tin_gate_w, tin_gate_b, false);

  // ---- 7/8. triangle attentions ----
  tri_attn(tas_ln_g, tas_ln_b, tas_feat_w, tas_q_w, tas_k_w, tas_v_w,
           tas_gate_w, tas_gate_b, tas_out_w, tas_out_b, true);
  tri_attn(tae_ln_g, tae_ln_b, tae_feat_w, tae_q_w, tae_k_w, tae_v_w,
           tae_gate_w, tae_gate_b, tae_out_w, tae_out_b, false);

  // ---- 9. pair transition ----
  transition(pair, MP, CZ, pt_ln_g, pt_ln_b, pt_w1, pt_b1, pt_w2, pt_b2);
}
